// AdaptiveAttention_18863496364213
// MI455X (gfx1250) — matrix operand order repaired and checked by reading, not compiled, not run on hardware
//
#include <hip/hip_runtime.h>
#include <hip/hip_bf16.h>
#include <math.h>

// ---------------------------------------------------------------------------
// Types for CDNA5 WMMA (wave32): v_wmma_f32_16x16x32_bf16
// ---------------------------------------------------------------------------
typedef __bf16 bf16_t;
typedef __attribute__((ext_vector_type(16))) __bf16 v16bf;
typedef __attribute__((ext_vector_type(8)))  __bf16 v8bf;
typedef __attribute__((ext_vector_type(2)))  __bf16 v2bf;
typedef __attribute__((ext_vector_type(8)))  float  v8f;

#define WMMA_BF16(a, b, c) \
    __builtin_amdgcn_wmma_f32_16x16x32_bf16(false, (a), false, (b), (short)0, (c), false, false)

// ---------------------------------------------------------------------------
// Optional CDNA5 async global->LDS copy (ASYNCcnt) with synchronous fallback.
// Builtin signature (from probe diagnostic): params are int4* in global/LDS AS.
// ---------------------------------------------------------------------------
#if defined(__has_builtin)
#  if __has_builtin(__builtin_amdgcn_global_load_async_to_lds_b128)
#    define ASYNC_LDS 1
#  endif
#  if __has_builtin(__builtin_amdgcn_s_wait_asynccnt)
#    define HAVE_WAIT_ASYNC_BUILTIN 1
#  endif
#endif

typedef int v4i_t __attribute__((vector_size(16)));
typedef __attribute__((address_space(1))) v4i_t* gv4i_p;
typedef __attribute__((address_space(3))) v4i_t* lv4i_p;

static __device__ __forceinline__ void copy16_to_lds(void* lds_dst, const void* gsrc) {
#ifdef ASYNC_LDS
    __builtin_amdgcn_global_load_async_to_lds_b128((gv4i_p)gsrc, (lv4i_p)lds_dst, 0, 0);
#else
    *(v8bf*)lds_dst = *(const v8bf*)gsrc;
#endif
}
static __device__ __forceinline__ void async_drain() {
#ifdef ASYNC_LDS
#  ifdef HAVE_WAIT_ASYNC_BUILTIN
    __builtin_amdgcn_s_wait_asynccnt(0);
#  else
    asm volatile("s_wait_asynccnt 0" ::: "memory");
#  endif
#endif
}

// A-fragment (16x32 bf16, MxK): source tile row-major [M][K], lda in elements.
static __device__ __forceinline__ v16bf load_a_frag(const bf16_t* tile, int lda, int m, int half) {
    union { v16bf v; v8bf h[2]; } u;
    const bf16_t* row = tile + m * lda + half * 8;
    u.h[0] = *(const v8bf*)(row);
    u.h[1] = *(const v8bf*)(row + 16);
    return u.v;
}

// B-fragment (32x16 bf16, KxN): source stored [N][K] (contiguous K), ldk in elements.
static __device__ __forceinline__ v16bf load_b_frag(const bf16_t* tileT, int ldk, int n, int half) {
    union { v16bf v; v8bf h[2]; } u;
    const bf16_t* row = tileT + n * ldk + half * 8;
    u.h[0] = *(const v8bf*)(row);
    u.h[1] = *(const v8bf*)(row + 16);
    return u.v;
}

// Problem constants
#define BB 4
#define SS 2048
#define DD 1024
#define HH 16
#define EE 64

// ---------------------------------------------------------------------------
// Kernel 1: softmax over head_selector[16] -> head_weights in ws
// ---------------------------------------------------------------------------
__global__ void head_softmax_kernel(const float* __restrict__ sel, float* __restrict__ hw) {
    int t = threadIdx.x;
    float x = (t < 16) ? sel[t] : -1e30f;
    float mx = x;
    #pragma unroll
    for (int msk = 1; msk < 16; msk <<= 1) mx = fmaxf(mx, __shfl_xor(mx, msk, 32));
    float e = __expf(x - mx);
    float s = e;
    #pragma unroll
    for (int msk = 1; msk < 16; msk <<= 1) s += __shfl_xor(s, msk, 32);
    if (t < 16) hw[t] = e / s;
}

// ---------------------------------------------------------------------------
// Kernel 2: per-(b,h) projection GEMM:  Out[b,h] = X[b] @ W[h] + bias[h]
//   M-tile 256, N = 64, K-step 32. 8 waves; each wave owns 32 rows (2 A-frags).
//   TRANSPOSE=1 stores [B,H,E,S] (for V^T), else [B,H,S,E].
// ---------------------------------------------------------------------------
template <int TRANSPOSE>
__global__ __launch_bounds__(256) void proj_kernel(
    const float* __restrict__ X, const float* __restrict__ W,
    const float* __restrict__ bias, bf16_t* __restrict__ Out)
{
    __shared__ __align__(16) bf16_t As[256 * 32];  // [m][k]
    __shared__ __align__(16) bf16_t Bt[64 * 32];   // [n=e][k]

    const int tid  = threadIdx.x;
    const int w    = tid >> 5;
    const int l    = tid & 31;
    const int half = l >> 4;
    const int lm   = l & 15;
    const int s0   = blockIdx.x * 256;
    const int h    = blockIdx.y;
    const int b    = blockIdx.z;
    const int m0   = w * 32;

    v8f acc[8];   // [mi*4 + nt]
    #pragma unroll
    for (int i = 0; i < 8; ++i)
        #pragma unroll
        for (int r = 0; r < 8; ++r) acc[i][r] = 0.0f;

    const float* Xb = X + ((size_t)b * SS + s0) * DD;
    const float* Wh = W + (size_t)h * DD * EE;

    for (int k0 = 0; k0 < DD; k0 += 32) {
        // A tile 256x32, f32 -> bf16, pair-packed stores
        #pragma unroll
        for (int it = 0; it < 16; ++it) {
            int li  = it * 256 + tid;          // pair index
            int row = li >> 4, cp = li & 15;
            float2 f = *(const float2*)(Xb + (size_t)row * DD + k0 + 2 * cp);
            v2bf t; t[0] = (bf16_t)f.x; t[1] = (bf16_t)f.y;
            *(v2bf*)(As + row * 32 + 2 * cp) = t;
        }
        // B tile 32x64 transposed into [e][k], pair-packed along k
        #pragma unroll
        for (int it = 0; it < 4; ++it) {
            int li = it * 256 + tid;           // pair index
            int kr = (li >> 6) * 2, e = li & 63;
            float f0 = Wh[(size_t)(k0 + kr) * EE + e];
            float f1 = Wh[(size_t)(k0 + kr + 1) * EE + e];
            v2bf t; t[0] = (bf16_t)f0; t[1] = (bf16_t)f1;
            *(v2bf*)(Bt + e * 32 + kr) = t;
        }
        __syncthreads();

        v16bf a0 = load_a_frag(As + m0 * 32,        32, lm, half);
        v16bf a1 = load_a_frag(As + (m0 + 16) * 32, 32, lm, half);
        #pragma unroll
        for (int nt = 0; nt < 4; ++nt) {
            v16bf bb = load_b_frag(Bt + nt * 16 * 32, 32, lm, half);
            acc[nt]     = WMMA_BF16(a0, bb, acc[nt]);
            acc[4 + nt] = WMMA_BF16(a1, bb, acc[4 + nt]);
        }
        __syncthreads();
    }

    const float* bh = bias + h * EE;
    #pragma unroll
    for (int mi = 0; mi < 2; ++mi)
        #pragma unroll
        for (int nt = 0; nt < 4; ++nt)
            #pragma unroll
            for (int r = 0; r < 8; ++r) {
                int srow = s0 + m0 + mi * 16 + r + 8 * half;
                int col  = nt * 16 + lm;
                float v  = acc[mi * 4 + nt][r] + bh[col];
                if (TRANSPOSE)
                    Out[(((size_t)b * HH + h) * EE + col) * SS + srow] = (bf16_t)v;
                else
                    Out[(((size_t)b * HH + h) * SS + srow) * EE + col] = (bf16_t)v;
            }
}

// ---------------------------------------------------------------------------
// Kernel 3: flash attention per (b,h), transposed-scores formulation.
//   S^T = K Q^T  : A = K tile [t][e] (LDS), B = Q [s][e] (B-frag from GLOBAL)
//   ctx^T = V^T P^T : A = V^T tile [e][t] (LDS), B = P^T [s][t] (wave LDS strip)
//   Each lane owns one query column s -> scalar online-softmax state,
//   reductions are in-lane + one shfl_xor(16).
// ---------------------------------------------------------------------------
__global__ __launch_bounds__(256) void attn_kernel(
    const bf16_t* __restrict__ Q, const bf16_t* __restrict__ K,
    const bf16_t* __restrict__ Vt, const float* __restrict__ hw,
    bf16_t* __restrict__ Comb)
{
    __shared__ __align__(16) bf16_t Kt[64 * 64];       // [t][e]
    __shared__ __align__(16) bf16_t Vs[64 * 64];       // [e][t]
    __shared__ __align__(16) bf16_t Ps[8 * 16 * 64];   // per-wave P^T strip [s][t]

    const int tid  = threadIdx.x;
    const int w    = tid >> 5;
    const int l    = tid & 31;
    const int half = l >> 4;
    const int lm   = l & 15;
    const int s0   = blockIdx.x * 128;
    const int h    = blockIdx.y;
    const int b    = blockIdx.z;
    const size_t bh = (size_t)b * HH + h;

    const bf16_t* Qb = Q  + bh * SS * EE;
    const bf16_t* Kb = K  + bh * SS * EE;
    const bf16_t* Vb = Vt + bh * EE * SS;

    const int sw = s0 + w * 16;            // this wave's 16 query rows
    // Q consumed directly as B-fragments from global ([s][e] == [N][K] layout)
    v16bf qb0 = load_b_frag(Qb + (size_t)sw * EE,      EE, lm, half);
    v16bf qb1 = load_b_frag(Qb + (size_t)sw * EE + 32, EE, lm, half);

    float mrun = -1e30f, lrun = 0.0f;      // per-lane: one query column s = sw+lm
    v8f ctxT[4];                           // [e][s] accumulators, et-tiles
    #pragma unroll
    for (int et = 0; et < 4; ++et)
        #pragma unroll
        for (int r = 0; r < 8; ++r) ctxT[et][r] = 0.0f;

    bf16_t* Pw = Ps + w * 16 * 64;
    const float scale = 0.125f;            // 1/sqrt(64)

    for (int t0 = 0; t0 < SS; t0 += 64) {
        // Stage K tile (contiguous 8KB) and V^T tile (64 rows, stride S) via async
        copy16_to_lds((v8bf*)Kt + tid,        (const v8bf*)(Kb + (size_t)t0 * EE) + tid);
        copy16_to_lds((v8bf*)Kt + tid + 256,  (const v8bf*)(Kb + (size_t)t0 * EE) + tid + 256);
        {
            int row = tid >> 2, seg = tid & 3;
            copy16_to_lds(Vs + row * 64 + seg * 16, Vb + (size_t)row * SS + t0 + seg * 16);
        }
        async_drain();
        __syncthreads();

        // S^T tiles: 4 mt-subtiles of K rows x 16 query cols, K-dim = E = 64
        v8f sc[4];
        #pragma unroll
        for (int mt = 0; mt < 4; ++mt) {
            #pragma unroll
            for (int r = 0; r < 8; ++r) sc[mt][r] = 0.0f;
            v16bf ka0 = load_a_frag(Kt + mt * 16 * 64,      64, lm, half);
            v16bf ka1 = load_a_frag(Kt + mt * 16 * 64 + 32, 64, lm, half);
            sc[mt] = WMMA_BF16(ka0, qb0, sc[mt]);
            sc[mt] = WMMA_BF16(ka1, qb1, sc[mt]);
        }

        // Online softmax over t: in-lane over 32 values + one cross-half shuffle
        float mx = -1e30f;
        #pragma unroll
        for (int mt = 0; mt < 4; ++mt)
            #pragma unroll
            for (int r = 0; r < 8; ++r) mx = fmaxf(mx, sc[mt][r]);
        mx = fmaxf(mx, __shfl_xor(mx, 16, 32));
        float mnew = fmaxf(mrun, mx * scale);
        float corr = __expf(mrun - mnew);

        float psum = 0.0f;
        #pragma unroll
        for (int mt = 0; mt < 4; ++mt) {
            v8bf pv;
            #pragma unroll
            for (int r = 0; r < 8; ++r) {
                float p = __expf(sc[mt][r] * scale - mnew);
                psum += p;
                pv[r] = (bf16_t)p;
            }
            // P^T strip [s][t]: lane's 8 t-values are consecutive -> one b128 store
            *(v8bf*)(Pw + lm * 64 + mt * 16 + 8 * half) = pv;
        }
        psum += __shfl_xor(psum, 16, 32);
        lrun = lrun * corr + psum;
        mrun = mnew;

        #pragma unroll
        for (int et = 0; et < 4; ++et)
            #pragma unroll
            for (int r = 0; r < 8; ++r) ctxT[et][r] *= corr;

        // ctx^T += V^T @ P^T  (wave-private strip; DS ops in-order per wave)
        v16bf pb0 = load_b_frag(Pw,      64, lm, half);
        v16bf pb1 = load_b_frag(Pw + 32, 64, lm, half);
        #pragma unroll
        for (int et = 0; et < 4; ++et) {
            v16bf va0 = load_a_frag(Vs + et * 16 * 64,      64, lm, half);
            v16bf va1 = load_a_frag(Vs + et * 16 * 64 + 32, 64, lm, half);
            ctxT[et] = WMMA_BF16(va0, pb0, ctxT[et]);
            ctxT[et] = WMMA_BF16(va1, pb1, ctxT[et]);
        }
        __syncthreads();
    }

    // Epilogue: ctx^T[e][s] -> Comb[b, s, h*64+e], 16B-packed stores
    const float inv = hw[h] / lrun;        // per-lane (its own s)
    const int srow = sw + lm;
    bf16_t* outp = Comb + (((size_t)b * SS + srow) * HH + h) * EE;
    #pragma unroll
    for (int et = 0; et < 4; ++et) {
        v8bf ov;
        #pragma unroll
        for (int r = 0; r < 8; ++r) ov[r] = (bf16_t)(ctxT[et][r] * inv);
        *(v8bf*)(outp + et * 16 + 8 * half) = ov;
    }
}

// ---------------------------------------------------------------------------
// Kernel 4: output GEMM:  Out = Comb(bf16 [B*S, 1024]) @ Wo(f32 [1024,1024]) + bo
//   M-tile 256, N-tile 64; 8 waves, 32 rows per wave.
// ---------------------------------------------------------------------------
__global__ __launch_bounds__(256) void out_gemm_kernel(
    const bf16_t* __restrict__ Acomb, const float* __restrict__ Wo,
    const float* __restrict__ bo, float* __restrict__ Out)
{
    constexpr int Kd = HH * EE;   // 1024
    constexpr int Nd = DD;        // 1024
    __shared__ __align__(16) bf16_t As[256 * 32];
    __shared__ __align__(16) bf16_t Bt[64 * 32];

    const int tid  = threadIdx.x;
    const int w    = tid >> 5;
    const int l    = tid & 31;
    const int half = l >> 4;
    const int lm   = l & 15;
    const int m0g  = blockIdx.x * 256;
    const int n0g  = blockIdx.y * 64;
    const int m0   = w * 32;

    v8f acc[8];
    #pragma unroll
    for (int i = 0; i < 8; ++i)
        #pragma unroll
        for (int r = 0; r < 8; ++r) acc[i][r] = 0.0f;

    for (int k0 = 0; k0 < Kd; k0 += 32) {
        // A tile 256x32: straight bf16 copy, one row per thread, async to LDS
        {
            const bf16_t* src = Acomb + (size_t)(m0g + tid) * Kd + k0;
            #pragma unroll
            for (int seg = 0; seg < 4; ++seg)
                copy16_to_lds(As + tid * 32 + seg * 8, src + seg * 8);
        }
        // B tile 32x64 transposed into [n][k], f32 -> bf16 pair-packed
        #pragma unroll
        for (int it = 0; it < 4; ++it) {
            int li = it * 256 + tid;
            int kr = (li >> 6) * 2, e = li & 63;
            float f0 = Wo[(size_t)(k0 + kr) * Nd + n0g + e];
            float f1 = Wo[(size_t)(k0 + kr + 1) * Nd + n0g + e];
            v2bf t; t[0] = (bf16_t)f0; t[1] = (bf16_t)f1;
            *(v2bf*)(Bt + e * 32 + kr) = t;
        }
        async_drain();
        __syncthreads();

        v16bf a0 = load_a_frag(As + m0 * 32,        32, lm, half);
        v16bf a1 = load_a_frag(As + (m0 + 16) * 32, 32, lm, half);
        #pragma unroll
        for (int nt = 0; nt < 4; ++nt) {
            v16bf bb = load_b_frag(Bt + nt * 16 * 32, 32, lm, half);
            acc[nt]     = WMMA_BF16(a0, bb, acc[nt]);
            acc[4 + nt] = WMMA_BF16(a1, bb, acc[4 + nt]);
        }
        __syncthreads();
    }

    #pragma unroll
    for (int mi = 0; mi < 2; ++mi)
        #pragma unroll
        for (int nt = 0; nt < 4; ++nt)
            #pragma unroll
            for (int r = 0; r < 8; ++r) {
                int row = m0g + m0 + mi * 16 + r + 8 * half;
                int col = n0g + nt * 16 + lm;
                Out[(size_t)row * Nd + col] = acc[mi * 4 + nt][r] + bo[col];
            }
}

// ---------------------------------------------------------------------------
// Host-side launch
// ---------------------------------------------------------------------------
extern "C" void kernel_launch(void* const* d_in, const int* in_sizes, int n_in,
                              void* d_out, int out_size, void* d_ws, size_t ws_size,
                              hipStream_t stream) {
    const float* query = (const float*)d_in[0];
    const float* key   = (const float*)d_in[1];
    const float* value = (const float*)d_in[2];
    const float* Wq    = (const float*)d_in[3];
    const float* bq    = (const float*)d_in[4];
    const float* Wk    = (const float*)d_in[5];
    const float* bk    = (const float*)d_in[6];
    const float* Wv    = (const float*)d_in[7];
    const float* bv    = (const float*)d_in[8];
    const float* Wo    = (const float*)d_in[9];
    const float* bo    = (const float*)d_in[10];
    const float* hsel  = (const float*)d_in[11];
    float* out = (float*)d_out;

    char* ws = (char*)d_ws;
    float* hw = (float*)ws;
    size_t off = 256;
    const size_t QSZ = (size_t)BB * HH * SS * EE * sizeof(bf16_t);  // 16 MB each
    bf16_t* Qws  = (bf16_t*)(ws + off); off += QSZ;
    bf16_t* Kws  = (bf16_t*)(ws + off); off += QSZ;
    bf16_t* Vtws = (bf16_t*)(ws + off); off += QSZ;
    bf16_t* Comb = (bf16_t*)(ws + off); off += QSZ;

    head_softmax_kernel<<<1, 32, 0, stream>>>(hsel, hw);

    dim3 gp(SS / 256, HH, BB);
    proj_kernel<0><<<gp, 256, 0, stream>>>(query, Wq, bq, Qws);
    proj_kernel<0><<<gp, 256, 0, stream>>>(key,   Wk, bk, Kws);
    proj_kernel<1><<<gp, 256, 0, stream>>>(value, Wv, bv, Vtws);   // store V^T

    attn_kernel<<<dim3(SS / 128, HH, BB), 256, 0, stream>>>(Qws, Kws, Vtws, hw, Comb);

    out_gemm_kernel<<<dim3((BB * SS) / 256, DD / 64), 256, 0, stream>>>(Comb, Wo, bo, out);
}
